// Model_12618613916306
// MI455X (gfx1250) — matrix operand order repaired and checked by reading, not compiled, not run on hardware
//
#include <hip/hip_runtime.h>
#include <hip/hip_bf16.h>

typedef __attribute__((ext_vector_type(16))) _Float16 v16h;
typedef __attribute__((ext_vector_type(8)))  _Float16 v8h;
typedef __attribute__((ext_vector_type(8)))  float    v8f;

#define N_USER_C  100000
#define N_MOVIE_C 50000
#define DHID      128

// ---------------------------------------------------------------------------
// zero a float region (n must be a multiple of 4)
__global__ void k_zero(float* __restrict__ p, long n) {
  long i = (long)blockIdx.x * blockDim.x + threadIdx.x;
  long stride = (long)gridDim.x * blockDim.x;
  long n4 = n >> 2;
  for (; i < n4; i += stride) {
    *(float4*)(p + i * 4) = make_float4(0.f, 0.f, 0.f, 0.f);
  }
}

// f32 -> f16 weight conversion
__global__ void k_cvt_f16(const float* __restrict__ s, _Float16* __restrict__ d, int n) {
  int i = blockIdx.x * blockDim.x + threadIdx.x;
  int st = gridDim.x * blockDim.x;
  for (; i < n; i += st) d[i] = (_Float16)s[i];
}

// ---------------------------------------------------------------------------
// Pass 1: both directions of the bipartite mean-aggregation in one edge sweep.
// One wave (32 lanes) per edge; lane l owns features [4l, 4l+4).
__global__ void k_edge_agg2(const int* __restrict__ eu, const int* __restrict__ em,
                            const float* __restrict__ xu, const float* __restrict__ xm,
                            float* __restrict__ agg_u, float* __restrict__ agg_m,
                            float* __restrict__ cnt_u, float* __restrict__ cnt_m, int E) {
  int lane = threadIdx.x & 31;
  long wid = ((long)blockIdx.x * blockDim.x + threadIdx.x) >> 5;
  long nw  = ((long)gridDim.x * blockDim.x) >> 5;
  int c = lane * 4;
  for (long e = wid; e < E; e += nw) {
    int u = eu[e], m = em[e];
    float4 vm = *(const float4*)(xm + (size_t)m * DHID + c);
    float4 vu = *(const float4*)(xu + (size_t)u * DHID + c);
    float* du = agg_u + (size_t)u * DHID + c;
    float* dm = agg_m + (size_t)m * DHID + c;
    atomicAdd(du + 0, vm.x); atomicAdd(du + 1, vm.y);
    atomicAdd(du + 2, vm.z); atomicAdd(du + 3, vm.w);
    atomicAdd(dm + 0, vu.x); atomicAdd(dm + 1, vu.y);
    atomicAdd(dm + 2, vu.z); atomicAdd(dm + 3, vu.w);
    if (lane == 0) { atomicAdd(cnt_u + u, 1.0f); atomicAdd(cnt_m + m, 1.0f); }
  }
}

// Pass 3: movie_x -> user aggregation only.
__global__ void k_edge_agg1(const int* __restrict__ eu, const int* __restrict__ em,
                            const float* __restrict__ xsrc, float* __restrict__ agg, int E) {
  int lane = threadIdx.x & 31;
  long wid = ((long)blockIdx.x * blockDim.x + threadIdx.x) >> 5;
  long nw  = ((long)gridDim.x * blockDim.x) >> 5;
  int c = lane * 4;
  for (long e = wid; e < E; e += nw) {
    int u = eu[e], m = em[e];
    float4 v = *(const float4*)(xsrc + (size_t)m * DHID + c);
    float* d = agg + (size_t)u * DHID + c;
    atomicAdd(d + 0, v.x); atomicAdd(d + 1, v.y);
    atomicAdd(d + 2, v.z); atomicAdd(d + 3, v.w);
  }
}

// ---------------------------------------------------------------------------
// Fragment-order LDS staging helpers.
// A-layout used here (16-bit, 16x32 per chunk, contiguous k order): element (m, k) lives in
//   lane = m + 16*g,  g = (k>>4)&1 within a 32-chunk,
//   half index h = k&15,  chunk kk = k>>5.
// Per-lane fragment storage: frag[lane*(K/2) + kk*16 + h] -> each lane's
// per-chunk fragment is 16 contiguous halves = one aligned v16h LDS load.
// An 8-element octet (k0 multiple of 8) is contiguous => one v8h store.
__device__ __forceinline__ int frag_base(int r, int c, int khalf /*K/2*/) {
  int kk = c >> 5, kr = c & 31;
  int g  = (kr >> 4) & 1;
  int h0 = kr & 8;
  return (r + 16 * g) * khalf + kk * 16 + h0;
}

// ---------------------------------------------------------------------------
// out = relu( (agg/max(cnt,1)) @ Wl.T + x @ Wr.T + bias )
// 256 threads = 8 wave32 per block; block owns 16 rows; wave w owns cols [16w,16w+16).
// B fragment for lane = 16 contiguous halves of W row n starting at k0+16g.
__global__ void k_sage_out(const float* __restrict__ agg, const float* __restrict__ cnt,
                           const _Float16* __restrict__ Wl,
                           const float* __restrict__ x, const _Float16* __restrict__ Wr,
                           const float* __restrict__ bias, float* __restrict__ out,
                           int do_relu) {
  __shared__ alignas(32) _Float16 AsF[32 * 64];
  __shared__ alignas(32) _Float16 XsF[32 * 64];
  int tid = threadIdx.x;
  int row0 = blockIdx.x * 16;
  int r  = tid >> 4;          // tile row 0..15
  int c0 = (tid & 15) * 8;    // feature octet
  int fb = frag_base(r, c0, 64);
  size_t src = (size_t)(row0 + r) * DHID + c0;
  float inv = 1.0f / fmaxf(cnt[row0 + r], 1.0f);
  v8h va, vx;
#pragma unroll
  for (int k = 0; k < 8; k++) {
    va[k] = (_Float16)(agg[src + k] * inv);
    vx[k] = (_Float16)x[src + k];
  }
  *(v8h*)&AsF[fb] = va;
  *(v8h*)&XsF[fb] = vx;
  __syncthreads();

  int lane = tid & 31, g = lane >> 4, ml = lane & 15;
  int n = (tid >> 5) * 16 + ml;
  const _Float16* wl = Wl + (size_t)n * DHID + 16 * g;
  const _Float16* wr = Wr + (size_t)n * DHID + 16 * g;
  const _Float16* af = &AsF[lane * 64];
  const _Float16* xf = &XsF[lane * 64];

  v8f c = {};
#pragma unroll
  for (int kc = 0; kc < 4; kc++) {
    v16h a  = *(const v16h*)(af + kc * 16);
    v16h b  = *(const v16h*)(wl + kc * 32);
    c = __builtin_amdgcn_wmma_f32_16x16x32_f16(false, a, false, b, (short)0, c, false, false);
    v16h a2 = *(const v16h*)(xf + kc * 16);
    v16h b2 = *(const v16h*)(wr + kc * 32);
    c = __builtin_amdgcn_wmma_f32_16x16x32_f16(false, a2, false, b2, (short)0, c, false, false);
  }

  float bv = bias[n];
#pragma unroll
  for (int j = 0; j < 8; j++) {
    float v = c[j] + bv;
    if (do_relu) v = fmaxf(v, 0.0f);
    out[(size_t)(row0 + j + 8 * g) * DHID + n] = v;
  }
}

// out = x @ W.T + bias      (single-matmul linear head, no relu)
__global__ void k_lin_out(const float* __restrict__ x, const _Float16* __restrict__ W,
                          const float* __restrict__ bias, float* __restrict__ out) {
  __shared__ alignas(32) _Float16 XsF[32 * 64];
  int tid = threadIdx.x;
  int row0 = blockIdx.x * 16;
  int r  = tid >> 4;
  int c0 = (tid & 15) * 8;
  int fb = frag_base(r, c0, 64);
  size_t src = (size_t)(row0 + r) * DHID + c0;
  v8h vx;
#pragma unroll
  for (int k = 0; k < 8; k++) vx[k] = (_Float16)x[src + k];
  *(v8h*)&XsF[fb] = vx;
  __syncthreads();

  int lane = tid & 31, g = lane >> 4, ml = lane & 15;
  int n = (tid >> 5) * 16 + ml;
  const _Float16* wp = W + (size_t)n * DHID + 16 * g;
  const _Float16* xf = &XsF[lane * 64];

  v8f c = {};
#pragma unroll
  for (int kc = 0; kc < 4; kc++) {
    v16h a = *(const v16h*)(xf + kc * 16);
    v16h b = *(const v16h*)(wp + kc * 32);
    c = __builtin_amdgcn_wmma_f32_16x16x32_f16(false, a, false, b, (short)0, c, false, false);
  }

  float bv = bias[n];
#pragma unroll
  for (int j = 0; j < 8; j++)
    out[(size_t)(row0 + j + 8 * g) * DHID + n] = c[j] + bv;
}

// ---------------------------------------------------------------------------
// Fused decoder: z = relu(concat(user_out[row], movie_out[col]) @ d1_W.T + d1_b)
//                out = z @ d2_W.T + d2_b            (one scalar per sample)
// 16 samples per block; d1 via WMMA over K=256; d2 matvec via __shfl_xor
// butterfly over 16-lane groups + LDS atomic combine across the 8 waves.
__global__ void k_decoder(const int* __restrict__ eu, const int* __restrict__ em,
                          const float* __restrict__ uo, const float* __restrict__ mo,
                          const _Float16* __restrict__ W1, const float* __restrict__ b1,
                          const float* __restrict__ W2, const float* __restrict__ b2,
                          float* __restrict__ out) {
  __shared__ alignas(32) _Float16 ZsF[32 * 128];
  __shared__ float zacc[16];
  int tid = threadIdx.x;
  int s0 = blockIdx.x * 16;
  int r  = tid >> 4;
  int c0 = (tid & 15) * 16;     // two octets per thread
  int ui = eu[s0 + r];
  int mi = em[s0 + r];
#pragma unroll
  for (int oo = 0; oo < 2; oo++) {
    int cc = c0 + oo * 8;
    int fb = frag_base(r, cc, 128);
    const float* srcp = (cc < 128) ? (uo + (size_t)ui * 128 + cc)
                                   : (mo + (size_t)mi * 128 + (cc - 128));
    v8h v;
#pragma unroll
    for (int k = 0; k < 8; k++) v[k] = (_Float16)srcp[k];
    *(v8h*)&ZsF[fb] = v;
  }
  if (tid < 16) zacc[tid] = 0.0f;
  __syncthreads();

  int lane = tid & 31, g = lane >> 4, ml = lane & 15;
  int n = (tid >> 5) * 16 + ml;
  const _Float16* wp = W1 + (size_t)n * 256 + 16 * g;
  const _Float16* zf = &ZsF[lane * 128];

  v8f c = {};
#pragma unroll
  for (int kc = 0; kc < 8; kc++) {
    v16h a = *(const v16h*)(zf + kc * 16);
    v16h b = *(const v16h*)(wp + kc * 32);
    c = __builtin_amdgcn_wmma_f32_16x16x32_f16(false, a, false, b, (short)0, c, false, false);
  }

  float bv = b1[n];
  float w2 = W2[n];
#pragma unroll
  for (int j = 0; j < 8; j++) {
    float z = fmaxf(c[j] + bv, 0.0f);
    float v = z * w2;
    v += __shfl_xor(v, 1);
    v += __shfl_xor(v, 2);
    v += __shfl_xor(v, 4);
    v += __shfl_xor(v, 8);
    if (ml == 0) atomicAdd(&zacc[j + 8 * g], v);
  }
  __syncthreads();
  if (tid < 16) out[s0 + tid] = zacc[tid] + b2[0];
}

// ---------------------------------------------------------------------------
extern "C" void kernel_launch(void* const* d_in, const int* in_sizes, int n_in,
                              void* d_out, int out_size, void* d_ws, size_t ws_size,
                              hipStream_t stream) {
  const float* x_user  = (const float*)d_in[0];
  const float* x_movie = (const float*)d_in[1];
  const int*   e_user  = (const int*)d_in[2];
  const int*   e_movie = (const int*)d_in[3];
  // d_in[4] = num_samples (device scalar); the output length is out_size.
  const float* c1_Wl = (const float*)d_in[5];
  const float* c1_bl = (const float*)d_in[6];
  const float* c1_Wr = (const float*)d_in[7];
  const float* c2_Wl = (const float*)d_in[8];
  const float* c2_bl = (const float*)d_in[9];
  const float* c2_Wr = (const float*)d_in[10];
  const float* c3_Wl = (const float*)d_in[11];
  const float* c3_bl = (const float*)d_in[12];
  const float* c3_Wr = (const float*)d_in[13];
  const float* l1_W  = (const float*)d_in[14];
  const float* l1_b  = (const float*)d_in[15];
  const float* l2_W  = (const float*)d_in[16];
  const float* l2_b  = (const float*)d_in[17];
  const float* d1_W  = (const float*)d_in[18];
  const float* d1_b  = (const float*)d_in[19];
  const float* d2_W  = (const float*)d_in[20];
  const float* d2_b  = (const float*)d_in[21];
  int E = in_sizes[2];

  // ---- workspace carve-up (floats) ----
  float* ws = (float*)d_ws;
  size_t o = 0;
  float* agg_u   = ws + o; o += (size_t)N_USER_C  * DHID;   // reused: agg3, then user_out
  float* agg_m   = ws + o; o += (size_t)N_MOVIE_C * DHID;   // reused: movie_out
  float* cnt_u   = ws + o; o += 100096;
  float* cnt_m   = ws + o; o += 50048;
  float* user_x  = ws + o; o += (size_t)N_USER_C  * DHID;   // conv3 runs in-place here
  float* movie_x = ws + o; o += (size_t)N_MOVIE_C * DHID;
  _Float16* h16 = (_Float16*)(ws + o);                      // 256B-aligned region
  _Float16* c1Wl16 = h16 + 0 * 16384;
  _Float16* c1Wr16 = h16 + 1 * 16384;
  _Float16* c2Wl16 = h16 + 2 * 16384;
  _Float16* c2Wr16 = h16 + 3 * 16384;
  _Float16* c3Wl16 = h16 + 4 * 16384;
  _Float16* c3Wr16 = h16 + 5 * 16384;
  _Float16* l1W16  = h16 + 6 * 16384;
  _Float16* l2W16  = h16 + 7 * 16384;
  _Float16* d1W16  = h16 + 8 * 16384;   // 128x256

  // zero accumulators + counts (contiguous region)
  long nz = (size_t)N_USER_C * DHID + (size_t)N_MOVIE_C * DHID + 100096 + 50048;
  k_zero<<<2048, 256, 0, stream>>>(agg_u, nz);

  // weights -> f16
  k_cvt_f16<<<64, 256, 0, stream>>>(c1_Wl, c1Wl16, 16384);
  k_cvt_f16<<<64, 256, 0, stream>>>(c1_Wr, c1Wr16, 16384);
  k_cvt_f16<<<64, 256, 0, stream>>>(c2_Wl, c2Wl16, 16384);
  k_cvt_f16<<<64, 256, 0, stream>>>(c2_Wr, c2Wr16, 16384);
  k_cvt_f16<<<64, 256, 0, stream>>>(c3_Wl, c3Wl16, 16384);
  k_cvt_f16<<<64, 256, 0, stream>>>(c3_Wr, c3Wr16, 16384);
  k_cvt_f16<<<64, 256, 0, stream>>>(l1_W,  l1W16,  16384);
  k_cvt_f16<<<64, 256, 0, stream>>>(l2_W,  l2W16,  16384);
  k_cvt_f16<<<128, 256, 0, stream>>>(d1_W, d1W16,  32768);

  // edge sweep 1: both aggregation directions + degree counts
  k_edge_agg2<<<4096, 256, 0, stream>>>(e_user, e_movie, x_user, x_movie,
                                        agg_u, agg_m, cnt_u, cnt_m, E);

  // conv1: user_x = relu(mean_agg @ c1_Wl.T + c1_bl + x_user @ c1_Wr.T)
  k_sage_out<<<N_USER_C / 16, 256, 0, stream>>>(agg_u, cnt_u, c1Wl16,
                                                x_user, c1Wr16, c1_bl, user_x, 1);
  // conv2: movie_x
  k_sage_out<<<N_MOVIE_C / 16, 256, 0, stream>>>(agg_m, cnt_m, c2Wl16,
                                                 x_movie, c2Wr16, c2_bl, movie_x, 1);

  // edge sweep 2: aggregate movie_x into users (reuse agg_u, re-zeroed)
  k_zero<<<2048, 256, 0, stream>>>(agg_u, (long)N_USER_C * DHID);
  k_edge_agg1<<<4096, 256, 0, stream>>>(e_user, e_movie, movie_x, agg_u, E);

  // conv3 (in-place on user_x)
  k_sage_out<<<N_USER_C / 16, 256, 0, stream>>>(agg_u, cnt_u, c3Wl16,
                                                user_x, c3Wr16, c3_bl, user_x, 1);

  // user_out = user_x @ l1_W.T + l1_b   (into agg_u)
  k_lin_out<<<N_USER_C / 16, 256, 0, stream>>>(user_x, l1W16, l1_b, agg_u);
  // movie_out = movie_x @ l2_W.T + l2_b (into agg_m)
  k_lin_out<<<N_MOVIE_C / 16, 256, 0, stream>>>(movie_x, l2W16, l2_b, agg_m);

  // fused edge decoder -> d_out
  k_decoder<<<out_size / 16, 256, 0, stream>>>(e_user, e_movie, agg_u, agg_m,
                                               d1W16, d1_b, d2_W, d2_b, (float*)d_out);
}
